// MTFLayer_81655918232152
// MI455X (gfx1250) — matrix operand order repaired and checked by reading, not compiled, not run on hardware
//
#include <hip/hip_runtime.h>

typedef __attribute__((ext_vector_type(16))) _Float16 v16h;
typedef __attribute__((ext_vector_type(8)))  _Float16 v8h;
typedef __attribute__((ext_vector_type(8)))  float    v8f;

#define NBINS 50
#define KPAD  64
#define WLEN  128
#define LO   (-1.0f)
#define HI   ( 1.0f)
#define EPSV ( 1e-6f)

#define SHUF16(a, b) __builtin_shufflevector((a), (b), 0,1,2,3,4,5,6,7,8,9,10,11,12,13,14,15)

// One workgroup (128 threads = 4 wave32) per series.
// MTF[m,n] = Hc[bin[m],bin[n]] / rowsum[bin[m]] computed as a one-hot GEMM:
//   A[m,k] = Hh[bin[m]][k]  (f16 counts <=127, exact), B[k,n] = Eh one-hot,
//   f32 accumulate via v_wmma_f32_16x16x32_f16, scale by 1/rowsum at store.
// Operands are staged in LDS as f16 tables so fragments load as ds_load_b128.
__global__ __launch_bounds__(128) void mtf_wmma_kernel(const float* __restrict__ X,
                                                       float* __restrict__ out) {
    __shared__ __align__(16) float    Hc[NBINS * NBINS];      // f32 integer counts
    __shared__ __align__(16) _Float16 Hh[NBINS * KPAD];       // f16 counts, K padded to 64
    __shared__ __align__(16) _Float16 Eh[WLEN * KPAD];        // f16 one-hot rows
    __shared__ int   binsS[WLEN];
    __shared__ float rowInv[NBINS];
    __shared__ float mscale[WLEN];
    __shared__ float wred[8];

    const int tid  = threadIdx.x;      // 0..127
    const int wave = tid >> 5;         // 0..3
    const int lane = tid & 31;
    const int ser  = blockIdx.x;

    // ---- load one sample per thread, wave32 min/max reduction ----
    float x  = X[(size_t)ser * WLEN + tid];
    float mn = x, mx = x;
    #pragma unroll
    for (int off = 16; off >= 1; off >>= 1) {
        mn = fminf(mn, __shfl_xor(mn, off, 32));
        mx = fmaxf(mx, __shfl_xor(mx, off, 32));
    }
    if (lane == 0) { wred[wave] = mn; wred[4 + wave] = mx; }
    __syncthreads();
    mn = fminf(fminf(wred[0], wred[1]), fminf(wred[2], wred[3]));
    mx = fmaxf(fmaxf(wred[4], wred[5]), fmaxf(wred[6], wred[7]));

    // ---- min-max scale + bucketize (searchsorted side='left') ----
    const float step = (HI - LO) / (float)NBINS;
    const float xs   = (x - mn) / (mx - mn + EPSV) * (HI - LO) + LO;
    int b = 0;
    #pragma unroll
    for (int k = 1; k <= NBINS - 1; ++k)
        b += (LO + step * (float)k < xs) ? 1 : 0;   // count boundaries < xs
    binsS[tid] = b;

    // ---- one-hot row for this thread's bin (zero 128B, poke one half) ----
    {
        unsigned int* row = (unsigned int*)&Eh[tid * KPAD];
        #pragma unroll
        for (int j = 0; j < KPAD / 2; ++j) row[j] = 0u;
        row[b >> 1] = (b & 1) ? 0x3C000000u : 0x00003C00u;   // f16 1.0
    }

    // ---- zero histogram ----
    #pragma unroll
    for (int i = tid; i < NBINS * NBINS; i += WLEN) Hc[i] = 0.0f;
    __syncthreads();

    // ---- transition counts (127 LDS float atomics, exact integers) ----
    if (tid < WLEN - 1)
        atomicAdd(&Hc[binsS[tid] * NBINS + binsS[tid + 1]], 1.0f);
    __syncthreads();

    // ---- f16 count table (padded) + row sums -> reciprocals ----
    for (int i = tid; i < NBINS * (KPAD / 2); i += WLEN) {
        const int r  = i >> 5;            // row 0..49
        const int c2 = (i & 31) * 2;      // col 0,2,..,62
        _Float16 f0 = (c2     < NBINS) ? (_Float16)Hc[r * NBINS + c2]     : (_Float16)0.0f;
        _Float16 f1 = (c2 + 1 < NBINS) ? (_Float16)Hc[r * NBINS + c2 + 1] : (_Float16)0.0f;
        Hh[r * KPAD + c2]     = f0;
        Hh[r * KPAD + c2 + 1] = f1;
    }
    if (tid < NBINS) {
        float rs = 0.0f;
        #pragma unroll 5
        for (int j = 0; j < NBINS; ++j) rs += Hc[tid * NBINS + j];
        rowInv[tid] = (rs > 0.0f) ? (1.0f / rs) : 1.0f;
    }
    __syncthreads();
    mscale[tid] = rowInv[binsS[tid]];
    __syncthreads();

    // ---- WMMA one-hot gather: wave w owns M-tiles {2w, 2w+1}, all 8 N-tiles ----
    const int h = lane >> 4;     // half-wave select
    const int l = lane & 15;
    float* outs = out + (size_t)ser * WLEN * WLEN;

    // Preload A fragments (2 M-tiles x 2 k-steps) and per-row scales.
    v16h  Af[2][2];
    float msc[2][8];
    #pragma unroll
    for (int mi = 0; mi < 2; ++mi) {
        const int m_base = (wave * 2 + mi) * 16;
        const int bm     = binsS[m_base + l];
        const v8h* Ar = (const v8h*)&Hh[bm * KPAD];
        // ISA 16-bit A layout: halves K = 8h..8h+7, 16+8h..16+8h+7 (then +32)
        Af[mi][0] = SHUF16(Ar[h],     Ar[2 + h]);
        Af[mi][1] = SHUF16(Ar[4 + h], Ar[6 + h]);
        #pragma unroll
        for (int r = 0; r < 8; ++r) msc[mi][r] = mscale[m_base + r + 8 * h];
    }

    #pragma unroll
    for (int nt = 0; nt < 8; ++nt) {
        const int n = nt * 16 + l;
        const v8h* Br = (const v8h*)&Eh[n * KPAD];
        // B layout matching A: halves K = 8h..8h+7, 16+8h..16+8h+7 (then +32)
        v16h B0 = SHUF16(Br[h],     Br[2 + h]);
        v16h B1 = SHUF16(Br[4 + h], Br[6 + h]);

        #pragma unroll
        for (int mi = 0; mi < 2; ++mi) {
            v8f acc = {};
            acc = __builtin_amdgcn_wmma_f32_16x16x32_f16(false, Af[mi][0], false, B0,
                                                         (short)0, acc, false, false);
            acc = __builtin_amdgcn_wmma_f32_16x16x32_f16(false, Af[mi][1], false, B1,
                                                         (short)0, acc, false, false);
            // C/D layout: VGPR r -> M = m_base + r + 8h, N = nt*16 + l
            float* po = outs + (size_t)((wave * 2 + mi) * 16 + 8 * h) * WLEN + n;
            #pragma unroll
            for (int r = 0; r < 8; ++r)
                po[(size_t)r * WLEN] = acc[r] * msc[mi][r];
        }
    }
}

extern "C" void kernel_launch(void* const* d_in, const int* in_sizes, int n_in,
                              void* d_out, int out_size, void* d_ws, size_t ws_size,
                              hipStream_t stream) {
    const float* X   = (const float*)d_in[0];
    float*       out = (float*)d_out;
    const int S = in_sizes[0] / WLEN;   // 256*6 = 1536 series
    mtf_wmma_kernel<<<dim3(S), dim3(128), 0, stream>>>(X, out);
}
